// Head_57088705298784
// MI455X (gfx1250) — matrix operand order repaired and checked by reading, not compiled, not run on hardware
//
#include <hip/hip_runtime.h>
#include <hip/hip_bf16.h>

// ---- CDNA5 WMMA vector types (match probe-confirmed builtin signatures) ----
typedef __attribute__((ext_vector_type(16))) __bf16 v16bf;
typedef __attribute__((ext_vector_type(8)))  __bf16 v8bf;
typedef __attribute__((ext_vector_type(8)))  float  v8f;
typedef __attribute__((ext_vector_type(4)))  float  v4f;
typedef __attribute__((ext_vector_type(4)))  int    v4i;

// Address-space-qualified v4i for the async global->LDS builtin
// (param 0 expects AS1 "int __vector(4) __device__ *" per hipcc diagnostic).
typedef __attribute__((address_space(1))) v4i gas_v4i;
typedef __attribute__((address_space(3))) v4i las_v4i;

#define NB     16
#define T_SEQ  2048
#define E_DIM  384
#define H_DIM  64
#define NEG_BIG (-1.0e30f)

static __device__ __forceinline__ v8f wmma_bf16(v16bf a, v16bf b, v8f c) {
    return __builtin_amdgcn_wmma_f32_16x16x32_bf16(
        /*neg_a=*/false, a, /*neg_b=*/false, b,
        /*c_mod=*/(short)0, c, /*reuse_a=*/false, /*reuse_b=*/false);
}

// ---- CDNA5 async global->LDS copy (ASYNCcnt path), with safe fallback ------
#if defined(__has_builtin)
#if __has_builtin(__builtin_amdgcn_global_load_async_to_lds_b128)
#define HAVE_ASYNC_LDS 1
#endif
#endif

template <int OFF>
static __device__ __forceinline__ void async_b128(const __bf16* g, __bf16* l) {
#ifdef HAVE_ASYNC_LDS
    // ISA: INST_OFFSET is added to BOTH the global and LDS addresses.
    __builtin_amdgcn_global_load_async_to_lds_b128(
        (gas_v4i*)const_cast<__bf16*>(g),
        (las_v4i*)l, OFF, 0);
#else
    *(v8bf*)((char*)l + OFF) = *(const v8bf*)((const char*)g + OFF);
#endif
}

static __device__ __forceinline__ void wait_async0() {
#ifdef HAVE_ASYNC_LDS
#if __has_builtin(__builtin_amdgcn_s_wait_asynccnt)
    __builtin_amdgcn_s_wait_asynccnt(0);
#else
    asm volatile("s_wait_asynccnt 0" ::: "memory");
#endif
#endif
}

// ============================================================================
// Kernel 1: qkv = x @ W_qkv  (f32 in, bf16 out; Q pre-scaled by 1/sqrt(H))
//   Q -> Qb [NB*T, 64] row-major
//   K -> Kb [NB*T, 64] row-major
//   V -> Vt [NB, 64, T]  (transposed so attention B-operands are contiguous)
// One wave computes one 16x16 output tile; 12 bf16 WMMAs over K=384.
// ============================================================================
__global__ __launch_bounds__(256) void qkv_proj_kernel(
        const float* __restrict__ x,     // [NB*T, E]
        const float* __restrict__ W,     // [E, 192]
        __bf16* __restrict__ Qb,
        __bf16* __restrict__ Kb,
        __bf16* __restrict__ Vt)
{
    const int lane = threadIdx.x & 31;
    const int wave = threadIdx.x >> 5;
    const int tile = blockIdx.x * 8 + wave;      // 2048*12 tiles total
    const int rowTile = tile / 12;
    const int colTile = tile % 12;
    const int m0 = rowTile * 16;
    const int n0 = colTile * 16;                 // uniform per wave
    const int half = lane >> 4;                  // lane group (0: lanes 0-15)
    const int l16  = lane & 15;

    const float* xrow = x + (size_t)(m0 + l16) * E_DIM;
    v8f acc = {};

    for (int k0 = 0; k0 < E_DIM; k0 += 32) {
        // ---- A tile 16x32 (bf16). Per lane: K = off..off+15
        const int offA = k0 + (half ? 16 : 0);
        v4f a0 = *(const v4f*)(xrow + offA);
        v4f a1 = *(const v4f*)(xrow + offA + 4);
        v4f a2 = *(const v4f*)(xrow + offA + 8);
        v4f a3 = *(const v4f*)(xrow + offA + 12);
        v16bf a;
#pragma unroll
        for (int j = 0; j < 4; ++j) {
            a[j]      = (__bf16)a0[j];
            a[4 + j]  = (__bf16)a1[j];
            a[8 + j]  = (__bf16)a2[j];
            a[12 + j] = (__bf16)a3[j];
        }
        // ---- B tile 32x16: per lane column n0+l16, K = kb..kb+15 (gather, L2-hot)
        const int kb = k0 + (half ? 16 : 0);
        const float* wcol = W + (size_t)kb * 192 + (n0 + l16);
        v16bf bm;
#pragma unroll
        for (int e = 0; e < 16; ++e) bm[e] = (__bf16)wcol[(size_t)e * 192];

        acc = wmma_bf16(a, bm, acc);
    }

    // ---- Store C tile. Branch on uniform n0 (one arm per wave).
    const int mBase = m0 + half * 8;
    if (n0 < 64) {                                   // Q, fold in 1/sqrt(64)
        __bf16* dst = Qb + (size_t)mBase * H_DIM + n0 + l16;
#pragma unroll
        for (int r = 0; r < 8; ++r)
            dst[(size_t)r * H_DIM] = (__bf16)(acc[r] * 0.125f);
    } else if (n0 < 128) {                           // K
        __bf16* dst = Kb + (size_t)mBase * H_DIM + (n0 - 64) + l16;
#pragma unroll
        for (int r = 0; r < 8; ++r)
            dst[(size_t)r * H_DIM] = (__bf16)acc[r];
    } else {                                         // V, transposed [NB,64,T]
        const int b = mBase >> 11;                   // row tiles never cross batch
        const int t = mBase & 2047;
        __bf16* dst = Vt + ((size_t)(b * H_DIM + (n0 - 128) + l16)) * T_SEQ + t;
#pragma unroll
        for (int r = 0; r < 8; ++r)
            dst[r] = (__bf16)acc[r];
    }
}

// ============================================================================
// Kernel 2: fused causal flash attention.
//   Block = 4 waves = 4 consecutive 16-query tiles of one batch.
//   K/V tiles (32 keys) cooperatively staged into double-buffered LDS with
//   CDNA5 async global->LDS copies, pipelined one tile ahead of the WMMAs.
//   One s_wait_asynccnt + one barrier per iteration. Online softmax in f32.
// ============================================================================
__global__ __launch_bounds__(128) void flash_attn_kernel(
        const __bf16* __restrict__ Qb,   // [NB*T, 64]
        const __bf16* __restrict__ Kb,   // [NB*T, 64]
        const __bf16* __restrict__ Vt,   // [NB, 64, T]
        float* __restrict__ out)         // [NB, T, 64]
{
    __shared__ __bf16 Kls[2][32 * 64];   // [key][dim]  : B-operand for Q@K^T
    __shared__ __bf16 Vls[2][64 * 32];   // [dim][key]  : B-operand for P@V
    __shared__ __bf16 Pls[4][16 * 32];   // per-wave P staging (C -> A layout)

    const int tid  = threadIdx.x;
    const int lane = tid & 31;
    const int wave = tid >> 5;
    const int b    = blockIdx.x >> 5;            // 32 blocks per batch
    const int qt0  = (blockIdx.x * 4) & 127;     // first q-tile of this block
    const int q0   = (qt0 + wave) * 16;          // this wave's query base
    const int half = lane >> 4;
    const int l16  = lane & 15;
    const int offA = half ? 16 : 0;
    const int offB = half ? 16 : 0;
    const size_t rowBase = (size_t)b * T_SEQ;

    // ---- Load the Q tile once as two A operands (K = 0..31 and 32..63)
    const __bf16* qrow = Qb + (rowBase + q0 + l16) * H_DIM;
    v16bf aQ0, aQ1;
    {
        v8bf t0 = *(const v8bf*)(qrow + offA);
        v8bf t1 = *(const v8bf*)(qrow + offA + 8);
        v8bf t2 = *(const v8bf*)(qrow + 32 + offA);
        v8bf t3 = *(const v8bf*)(qrow + 32 + offA + 8);
#pragma unroll
        for (int j = 0; j < 8; ++j) {
            aQ0[j] = t0[j]; aQ0[8 + j] = t1[j];
            aQ1[j] = t2[j]; aQ1[8 + j] = t3[j];
        }
    }

    float mrow[8], lrow[8];
#pragma unroll
    for (int r = 0; r < 8; ++r) { mrow[r] = NEG_BIG; lrow[r] = 0.0f; }
    v8f o0 = {}, o1 = {}, o2 = {}, o3 = {};

    __bf16* pls = &Pls[wave][0];
    const int kEnd      = q0 + 16;               // this wave's causal range
    const int kEndBlock = qt0 * 16 + 64;         // block max (uniform)

    // Cooperative-copy indices (128 threads, 32B each => 4KB per tile)
    const int kRow = tid >> 2, kSeg = tid & 3;   // K: row 0..31, dim seg 0..3
    const int vDim = tid >> 1, vSeg = tid & 1;   // V: dim 0..63, key seg 0..1
    const __bf16* kSrcBase = Kb + (rowBase + kRow) * H_DIM + kSeg * 16;
    const __bf16* vSrcBase = Vt + (size_t)(b * H_DIM + vDim) * T_SEQ + vSeg * 16;
    __bf16* kDst0 = &Kls[0][kRow * 64 + kSeg * 16];
    __bf16* vDst0 = &Vls[0][vDim * 32 + vSeg * 16];

    // ---- Prologue: stage tile 0 into buffer 0
    async_b128<0>(kSrcBase, kDst0);  async_b128<16>(kSrcBase, kDst0);
    async_b128<0>(vSrcBase, vDst0);  async_b128<16>(vSrcBase, vDst0);

    int buf = 0;
    for (int s0 = 0; s0 < kEndBlock; s0 += 32, buf ^= 1) {
        wait_async0();          // own async stores to LDS[buf] complete
        __syncthreads();        // everyone's complete -> LDS[buf] readable;
                                // also: all reads of LDS[buf^1] are done.

        // ---- Pipeline: issue async stage of the NEXT tile into LDS[buf^1]
        if (s0 + 32 < kEndBlock) {
            const __bf16* ks = kSrcBase + (size_t)(s0 + 32) * H_DIM;
            const __bf16* vs = vSrcBase + (s0 + 32);
            __bf16* kd = kDst0 + (buf ^ 1) * (32 * 64);
            __bf16* vd = vDst0 + (buf ^ 1) * (64 * 32);
            async_b128<0>(ks, kd);  async_b128<16>(ks, kd);
            async_b128<0>(vs, vd);  async_b128<16>(vs, vd);
        }

        if (s0 < kEnd) {
            // ---- S = Q K^T : two 16x16 C tiles, K-dim 64 (2 WMMAs each)
            const __bf16* kr0 = &Kls[buf][l16 * 64 + offB];
            const __bf16* kr1 = kr0 + 16 * 64;
            v16bf b00 = *(const v16bf*)(kr0);
            v16bf b01 = *(const v16bf*)(kr0 + 32);
            v16bf b10 = *(const v16bf*)(kr1);
            v16bf b11 = *(const v16bf*)(kr1 + 32);
            v8f c0 = {}, c1 = {};
            c0 = wmma_bf16(aQ0, b00, c0);
            c0 = wmma_bf16(aQ1, b01, c0);
            c1 = wmma_bf16(aQ0, b10, c1);
            c1 = wmma_bf16(aQ1, b11, c1);

            // ---- Causal mask + online softmax (row r lives in lane half)
            const int key0 = s0 + l16;
            const int key1 = key0 + 16;
            float sc[8];
#pragma unroll
            for (int r = 0; r < 8; ++r) {
                const int rowl = r + half * 8;
                const int qg = q0 + rowl;
                float s0v = (key0 <= qg) ? c0[r] : NEG_BIG;
                float s1v = (key1 <= qg) ? c1[r] : NEG_BIG;
                float rmax = fmaxf(s0v, s1v);
                rmax = fmaxf(rmax, __shfl_xor(rmax, 1));
                rmax = fmaxf(rmax, __shfl_xor(rmax, 2));
                rmax = fmaxf(rmax, __shfl_xor(rmax, 4));
                rmax = fmaxf(rmax, __shfl_xor(rmax, 8));
                const float mnew  = fmaxf(mrow[r], rmax);
                const float scale = __expf(mrow[r] - mnew);
                const float p0 = __expf(s0v - mnew);
                const float p1 = __expf(s1v - mnew);
                float rsum = p0 + p1;
                rsum += __shfl_xor(rsum, 1);
                rsum += __shfl_xor(rsum, 2);
                rsum += __shfl_xor(rsum, 4);
                rsum += __shfl_xor(rsum, 8);
                lrow[r] = lrow[r] * scale + rsum;
                mrow[r] = mnew;
                sc[r]   = scale;
                pls[rowl * 32 + l16]      = (__bf16)p0;
                pls[rowl * 32 + 16 + l16] = (__bf16)p1;
            }
#pragma unroll
            for (int r = 0; r < 8; ++r) {
                o0[r] *= sc[r]; o1[r] *= sc[r]; o2[r] *= sc[r]; o3[r] *= sc[r];
            }

            // ---- Reload P as A operand (same-wave DS RAW -> s_wait_dscnt)
            v16bf aP;
            {
                const __bf16* prow = pls + l16 * 32 + offA;
                v8bf t0 = *(const v8bf*)(prow);
                v8bf t1 = *(const v8bf*)(prow + 8);
#pragma unroll
                for (int j = 0; j < 8; ++j) { aP[j] = t0[j]; aP[8 + j] = t1[j]; }
            }

            // ---- O += P @ V : B operand contiguous along keys in Vls
            const __bf16* vb = &Vls[buf][l16 * 32 + offB];
            v16bf bv0 = *(const v16bf*)(vb);
            v16bf bv1 = *(const v16bf*)(vb + 16 * 32);
            v16bf bv2 = *(const v16bf*)(vb + 32 * 32);
            v16bf bv3 = *(const v16bf*)(vb + 48 * 32);
            o0 = wmma_bf16(aP, bv0, o0);
            o1 = wmma_bf16(aP, bv1, o1);
            o2 = wmma_bf16(aP, bv2, o2);
            o3 = wmma_bf16(aP, bv3, o3);
        }
    }

    // ---- Normalize and store (f32 output [NB, T, 64])
    float* orow = out + (rowBase + q0) * H_DIM;
#pragma unroll
    for (int r = 0; r < 8; ++r) {
        const float inv = 1.0f / lrow[r];
        const int rowl = r + half * 8;
        float* dst = orow + (size_t)rowl * H_DIM + l16;
        dst[0]  = o0[r] * inv;
        dst[16] = o1[r] * inv;
        dst[32] = o2[r] * inv;
        dst[48] = o3[r] * inv;
    }
}

// ============================================================================
extern "C" void kernel_launch(void* const* d_in, const int* in_sizes, int n_in,
                              void* d_out, int out_size, void* d_ws, size_t ws_size,
                              hipStream_t stream) {
    (void)in_sizes; (void)n_in; (void)out_size; (void)ws_size;
    const float* x = (const float*)d_in[0];      // [16, 2048, 384] f32
    const float* W = (const float*)d_in[1];      // [384, 192] f32
    float* out = (float*)d_out;                  // [16, 2048, 64] f32

    const size_t qkvElems = (size_t)NB * T_SEQ * H_DIM;   // 2M elems each
    __bf16* Qb = (__bf16*)d_ws;
    __bf16* Kb = Qb + qkvElems;
    __bf16* Vt = Kb + qkvElems;                  // total 12 MB of ws

    // 2048 row-tiles x 12 col-tiles = 24576 wave-tiles; 8 waves/block
    qkv_proj_kernel<<<3072, 256, 0, stream>>>(x, W, Qb, Kb, Vt);
    // 16 batches x 128 q-tiles = 2048 waves; 4 waves/block (shared K/V tiles)
    flash_attn_kernel<<<512, 128, 0, stream>>>(Qb, Kb, Vt, out);
}
